// MMMambaEncoderLayer_16638703305489
// MI455X (gfx1250) — matrix operand order repaired and checked by reading, not compiled, not run on hardware
//
#include <hip/hip_runtime.h>

// ---------------------------------------------------------------------------
// Mamba encoder layer for MI455X (gfx1250, wave32, WMMA).
// Pipeline: cvt->bf16 | in_proj(WMMA) | conv+silu | x_proj(WMMA) |
//           dt_proj+softplus | selective scan (f32, shfl-reduce) |
//           out_proj(WMMA) | layernorm+residual
//
// Roofline: 13.7 GFLOP of GEMMs, ~100 MB of intermediates -> entirely
// L2-resident (192 MB); HBM traffic ~ inputs+outputs (~25 MB ~ 1 us at
// 23.3 TB/s). bf16 WMMA (f32 accum) for GEMMs; f32 VALU for the serial
// selective scan and softplus (precision-sensitive recurrence).
// ---------------------------------------------------------------------------

typedef __attribute__((ext_vector_type(16))) __bf16 v16bf;
typedef __attribute__((ext_vector_type(8)))  __bf16 v8bf;
typedef __attribute__((ext_vector_type(8)))  float  v8f;

static constexpr int BN   = 2;
static constexpr int LN   = 2048;
static constexpr int DN   = 512;
static constexpr int DI   = 1024;   // d_inner
static constexpr int DS   = 16;     // d_state
static constexpr int DTR  = 32;     // dt_rank
static constexpr int MROW = BN * LN;  // 4096

// ---------------------------------------------------------------------------
// f32 -> bf16 conversion (weights + activations fed to WMMA)
// ---------------------------------------------------------------------------
__global__ __launch_bounds__(256) void cvt_f32_bf16(
    const float* __restrict__ src, __bf16* __restrict__ dst, int n)
{
    int i = blockIdx.x * 256 + threadIdx.x;
    if (i < n) dst[i] = (__bf16)src[i];
}

// ---------------------------------------------------------------------------
// Generic C[M,N](f32) = A[M,K](bf16) * W[N,K](bf16)^T  via v_wmma_f32_16x16x32_bf16
// Wave tile: 16M x 64N (A-frag reused across 4 B-frags). Block = 8 waves = 128M x 64N.
//
// A-frag layout (16-bit A 16x32): lane l -> m=l&15,
//   element e -> k = e + (l>>4)*16                  (one contiguous 16-elt run)
// B-frag layout (16-bit B 32x16): lane l -> n=l&15,
//   element e -> k = e + (l>>4)*16                  (one contiguous 16-elt run)
// C/D layout: vgpr j, lane l -> m = j + (l>>4)*8, n = l&15
// ---------------------------------------------------------------------------
__global__ __launch_bounds__(256) void gemm_bf16_nt(
    const __bf16* __restrict__ A, int lda,
    const __bf16* __restrict__ W, int ldw,
    float* __restrict__ C, int ldc,
    int M, int N, int K)
{
    const int lane = threadIdx.x & 31;
    const int wave = threadIdx.x >> 5;
    const int mbase = blockIdx.y * 128 + wave * 16;
    const int nbase = blockIdx.x * 64;
    if (mbase >= M || nbase >= N) return;

    const int col  = lane & 15;
    const int half = lane >> 4;

    const __bf16* arow = A + (size_t)(mbase + col) * lda + half * 16;
    const __bf16* w0   = W + (size_t)(nbase + col) * ldw + half * 16;
    const __bf16* w1   = w0 + (size_t)16 * ldw;
    const __bf16* w2   = w0 + (size_t)32 * ldw;
    const __bf16* w3   = w0 + (size_t)48 * ldw;

    v8f acc0 = {}; v8f acc1 = {}; v8f acc2 = {}; v8f acc3 = {};

    for (int k0 = 0; k0 < K; k0 += 32) {
        v8bf alo = *(const v8bf*)(arow + k0);        // k0+half*16 .. +7
        v8bf ahi = *(const v8bf*)(arow + k0 + 8);    // k0+half*16+8 ..
        v16bf af;
#pragma unroll
        for (int i = 0; i < 8; ++i) { af[i] = alo[i]; af[i + 8] = ahi[i]; }

        v16bf b0 = *(const v16bf*)(w0 + k0);
        v16bf b1 = *(const v16bf*)(w1 + k0);
        v16bf b2 = *(const v16bf*)(w2 + k0);
        v16bf b3 = *(const v16bf*)(w3 + k0);

        acc0 = __builtin_amdgcn_wmma_f32_16x16x32_bf16(false, af, false, b0, (short)0, acc0, false, false);
        acc1 = __builtin_amdgcn_wmma_f32_16x16x32_bf16(false, af, false, b1, (short)0, acc1, false, false);
        acc2 = __builtin_amdgcn_wmma_f32_16x16x32_bf16(false, af, false, b2, (short)0, acc2, false, false);
        acc3 = __builtin_amdgcn_wmma_f32_16x16x32_bf16(false, af, false, b3, (short)0, acc3, false, false);
    }

#pragma unroll
    for (int j = 0; j < 8; ++j) {
        float* crow = C + (size_t)(mbase + half * 8 + j) * ldc + nbase + col;
        crow[0]  = acc0[j];
        crow[16] = acc1[j];
        crow[32] = acc2[j];
        crow[48] = acc3[j];
    }
}

// ---------------------------------------------------------------------------
// Causal depthwise conv1d (k=4) + bias + SiLU on the x half of xz.
// xz layout: [m=b*L+t][2048] with xm = [:,0:1024], z = [:,1024:2048]
// out[t,d] = silu( b[d] + sum_{j=0..3} w[d,j] * xm[t-3+j, d] )
// ---------------------------------------------------------------------------
__global__ __launch_bounds__(256) void conv_silu(
    const float* __restrict__ xz, const float* __restrict__ cw,
    const float* __restrict__ cb,
    float* __restrict__ xmf, __bf16* __restrict__ xmb)
{
    int idx = blockIdx.x * 256 + threadIdx.x;   // m*1024 + d
    int d = idx & (DI - 1);
    int m = idx >> 10;
    int t = m & (LN - 1);
    const float* p = xz + (size_t)m * (2 * DI) + d;
    float w0 = cw[d * 4 + 0], w1 = cw[d * 4 + 1];
    float w2 = cw[d * 4 + 2], w3 = cw[d * 4 + 3];
    float acc = cb[d] + w3 * p[0];
    if (t >= 1) acc += w2 * p[-(2 * DI)];
    if (t >= 2) acc += w1 * p[-(4 * DI)];
    if (t >= 3) acc += w0 * p[-(6 * DI)];
    float s = acc / (1.f + expf(-acc));          // silu
    xmf[idx] = s;
    xmb[idx] = (__bf16)s;
}

// ---------------------------------------------------------------------------
// delta[m,d] = softplus( dbc[m,0:32] . dt_proj_w[d,0:32] + dt_proj_b[d] )
// K=32 stays f32 VALU (softplus argument is precision sensitive).
// ---------------------------------------------------------------------------
__global__ __launch_bounds__(256) void dtproj_softplus(
    const float* __restrict__ dbc, const float* __restrict__ dtw,
    const float* __restrict__ dtb, float* __restrict__ delta)
{
    int idx = blockIdx.x * 256 + threadIdx.x;   // m*1024 + d
    int d = idx & (DI - 1);
    int m = idx >> 10;
    const float4* dt4 = (const float4*)(dbc + (size_t)m * 64);
    const float4* w4  = (const float4*)(dtw + (size_t)d * DTR);
    float acc = dtb[d];
#pragma unroll
    for (int r = 0; r < DTR / 4; ++r) {
        float4 a = dt4[r]; float4 w = w4[r];
        acc += a.x * w.x + a.y * w.y + a.z * w.z + a.w * w.w;
    }
    // numerically stable softplus: max(x,0) + log1p(exp(-|x|))
    delta[idx] = fmaxf(acc, 0.f) + log1pf(expf(-fabsf(acc)));
}

// ---------------------------------------------------------------------------
// Selective scan, fused with D-skip and SiLU(z) gating.
// Block: (batch b, 64 d-channels); 1024 threads = 64 d x 16 states.
// Each lane owns h[d,s]; y = sum_s h*C via 4x shfl_xor inside 16-lane group.
// dA/dBx computed on the fly (never materialized: would be 2x268 MB).
// #pragma unroll 4 lets the h-independent loads of future timesteps issue
// ahead of the serial VALU recurrence chain.
// ---------------------------------------------------------------------------
__global__ __launch_bounds__(1024) void selective_scan(
    const float* __restrict__ delta, const float* __restrict__ xmf,
    const float* __restrict__ dbc,   const float* __restrict__ xz,
    const float* __restrict__ A_log, const float* __restrict__ Dsk,
    __bf16* __restrict__ yb)
{
    const int b = blockIdx.y;
    const int d = blockIdx.x * 64 + (threadIdx.x >> 4);
    const int s = threadIdx.x & 15;
    const float Ads = -expf(A_log[d * DS + s]);   // A = -exp(A_log)
    const float Dd  = Dsk[d];
    float h = 0.f;
    const size_t mb = (size_t)b * LN;
#pragma unroll 4
    for (int t = 0; t < LN; ++t) {
        const size_t m = mb + t;
        float dlt = delta[m * DI + d];
        float xv  = xmf [m * DI + d];
        float Bv  = dbc [m * 64 + DTR + s];        // B slice
        float Cv  = dbc [m * 64 + DTR + DS + s];   // C slice
        float dA  = expf(dlt * Ads);
        h = dA * h + (dlt * Bv) * xv;
        float yv = h * Cv;
        yv += __shfl_xor(yv, 1, 32);
        yv += __shfl_xor(yv, 2, 32);
        yv += __shfl_xor(yv, 4, 32);
        yv += __shfl_xor(yv, 8, 32);
        if (s == 0) {
            float z = xz[m * (2 * DI) + DI + d];
            float g = z / (1.f + expf(-z));        // silu(z)
            yb[m * DI + d] = (__bf16)((yv + xv * Dd) * g);
        }
    }
}

// ---------------------------------------------------------------------------
// out[m,:] = x[m,:] + layernorm(out1[m,:]) * ln_w + ln_b     (row = 512)
// One 256-thread block per row; 2 elements/thread; shfl + LDS reduction.
// ---------------------------------------------------------------------------
__global__ __launch_bounds__(256) void ln_residual(
    const float* __restrict__ x, const float* __restrict__ o1,
    const float* __restrict__ w, const float* __restrict__ bia,
    float* __restrict__ out)
{
    __shared__ float red[8];
    const int m   = blockIdx.x;
    const int tid = threadIdx.x;
    const int lane = tid & 31, wv = tid >> 5;
    const float* row = o1 + (size_t)m * DN;
    float v0 = row[tid], v1 = row[tid + 256];

    auto bsum = [&](float v) -> float {
        v += __shfl_xor(v, 16, 32); v += __shfl_xor(v, 8, 32);
        v += __shfl_xor(v, 4, 32);  v += __shfl_xor(v, 2, 32);
        v += __shfl_xor(v, 1, 32);
        if (lane == 0) red[wv] = v;
        __syncthreads();
        float t = (lane < 8) ? red[lane] : 0.f;
        t += __shfl_xor(t, 4, 32); t += __shfl_xor(t, 2, 32); t += __shfl_xor(t, 1, 32);
        t = __shfl(t, 0, 32);
        __syncthreads();
        return t;
    };

    float mu = bsum(v0 + v1) * (1.f / DN);
    float d0 = v0 - mu, d1 = v1 - mu;
    float var = bsum(d0 * d0 + d1 * d1) * (1.f / DN);
    float rs = rsqrtf(var + 1e-6f);
    const float* xr = x + (size_t)m * DN;
    out[(size_t)m * DN + tid]       = xr[tid]       + d0 * rs * w[tid]       + bia[tid];
    out[(size_t)m * DN + tid + 256] = xr[tid + 256] + d1 * rs * w[tid + 256] + bia[tid + 256];
}

// ---------------------------------------------------------------------------
extern "C" void kernel_launch(void* const* d_in, const int* in_sizes, int n_in,
                              void* d_out, int out_size, void* d_ws, size_t ws_size,
                              hipStream_t stream)
{
    const float* x      = (const float*)d_in[0];
    const float* w_in   = (const float*)d_in[1];
    const float* conv_w = (const float*)d_in[2];
    const float* conv_b = (const float*)d_in[3];
    const float* w_xp   = (const float*)d_in[4];
    const float* w_dt   = (const float*)d_in[5];
    const float* b_dt   = (const float*)d_in[6];
    const float* A_log  = (const float*)d_in[7];
    const float* Dsk    = (const float*)d_in[8];
    const float* w_out  = (const float*)d_in[9];
    const float* lnw    = (const float*)d_in[10];
    const float* lnb    = (const float*)d_in[11];
    float* out = (float*)d_out;

    // workspace carve-up (all intermediates fit in the 192 MB L2)
    char* ws = (char*)d_ws;
    auto carve = [&](size_t bytes) -> char* {
        char* p = ws; ws += (bytes + 255) & ~(size_t)255; return p;
    };
    float*  xz    = (float*) carve((size_t)MROW * 2 * DI * 4);   // 33.5 MB
    float*  dbc   = (float*) carve((size_t)MROW * 64 * 4);       //  1.0 MB
    float*  delta = (float*) carve((size_t)MROW * DI * 4);       // 16.8 MB
    float*  xmf   = (float*) carve((size_t)MROW * DI * 4);       // 16.8 MB
    float*  out1  = (float*) carve((size_t)MROW * DN * 4);       //  8.4 MB
    __bf16* xb    = (__bf16*)carve((size_t)MROW * DN * 2);       //  4.2 MB
    __bf16* winb  = (__bf16*)carve((size_t)2 * DI * DN * 2);     //  2.1 MB
    __bf16* xmb   = (__bf16*)carve((size_t)MROW * DI * 2);       //  8.4 MB
    __bf16* wxpb  = (__bf16*)carve((size_t)64 * DI * 2);         //  0.1 MB
    __bf16* yb    = (__bf16*)carve((size_t)MROW * DI * 2);       //  8.4 MB
    __bf16* woutb = (__bf16*)carve((size_t)DN * DI * 2);         //  1.0 MB

    // 1) bf16 conversions for WMMA operands
    cvt_f32_bf16<<<(MROW * DN) / 256, 256, 0, stream>>>(x, xb, MROW * DN);
    cvt_f32_bf16<<<(2 * DI * DN) / 256, 256, 0, stream>>>(w_in, winb, 2 * DI * DN);
    cvt_f32_bf16<<<(64 * DI) / 256, 256, 0, stream>>>(w_xp, wxpb, 64 * DI);
    cvt_f32_bf16<<<(DN * DI) / 256, 256, 0, stream>>>(w_out, woutb, DN * DI);

    // 2) in_proj: xz[4096,2048] = xb[4096,512] @ winb[2048,512]^T
    gemm_bf16_nt<<<dim3((2 * DI) / 64, MROW / 128), 256, 0, stream>>>(
        xb, DN, winb, DN, xz, 2 * DI, MROW, 2 * DI, DN);

    // 3) causal conv + SiLU -> xm (f32 for scan, bf16 for x_proj GEMM)
    conv_silu<<<(MROW * DI) / 256, 256, 0, stream>>>(xz, conv_w, conv_b, xmf, xmb);

    // 4) x_proj: dbc[4096,64] = xmb[4096,1024] @ wxpb[64,1024]^T
    gemm_bf16_nt<<<dim3(64 / 64, MROW / 128), 256, 0, stream>>>(
        xmb, DI, wxpb, DI, dbc, 64, MROW, 64, DI);

    // 5) dt_proj + softplus -> delta
    dtproj_softplus<<<(MROW * DI) / 256, 256, 0, stream>>>(dbc, w_dt, b_dt, delta);

    // 6) selective scan fused with D-skip and silu(z) gating -> yb (bf16)
    selective_scan<<<dim3(DI / 64, BN), 1024, 0, stream>>>(
        delta, xmf, dbc, xz, A_log, Dsk, yb);

    // 7) out_proj: out1[4096,512] = yb[4096,1024] @ woutb[512,1024]^T
    gemm_bf16_nt<<<dim3(DN / 64, MROW / 128), 256, 0, stream>>>(
        yb, DI, woutb, DI, out1, DN, MROW, DN, DI);

    // 8) residual + layernorm
    ln_residual<<<MROW, 256, 0, stream>>>(x, out1, lnw, lnb, out);
}
